// EncoderLayer_57595511439738
// MI455X (gfx1250) — matrix operand order repaired and checked by reading, not compiled, not run on hardware
//
#include <hip/hip_runtime.h>

typedef __attribute__((ext_vector_type(16))) __bf16   v16bf;
typedef __attribute__((ext_vector_type(8)))  float    v8f;
typedef __attribute__((ext_vector_type(4)))  unsigned u32x4;

union ABFrag { u32x4 q[2]; v16bf v; };

__device__ __forceinline__ unsigned f2bf1(float x) {
  unsigned u = __float_as_uint(x);
  return (u + 0x7FFFu + ((u >> 16) & 1u)) >> 16;   // RNE f32 -> bf16
}
__device__ __forceinline__ unsigned packbf(float lo, float hi) {
  return f2bf1(lo) | (f2bf1(hi) << 16);
}
__device__ __forceinline__ void atomicMaxF(float* addr, float val) {
  if (val >= 0.0f) atomicMax((int*)addr, __float_as_int(val));
  else             atomicMin((unsigned*)addr, __float_as_uint(val));
}

// ---------------------------------------------------------------- utilities
__global__ void fill_kernel(unsigned* __restrict__ p, unsigned val, int n) {
  int i = blockIdx.x * blockDim.x + threadIdx.x;
  if (i < n) p[i] = val;
}

__global__ void cvt_bf16_kernel(const float* __restrict__ x, unsigned* __restrict__ y, int npairs) {
  int i = blockIdx.x * blockDim.x + threadIdx.x;
  if (i < npairs) y[i] = packbf(x[2 * i], x[2 * i + 1]);
}

// Pack row-major f32 weight (K x N) into WMMA-B bf16 fragments:
// frag f = ks*(N/16)+nt, 256 dwords: lane (N=lane&15, Khalf=lane>>4), dword v holds K=2v,2v+1.
__global__ void pack_b_kernel(const float* __restrict__ W, unsigned* __restrict__ Bp, int K, int Nn) {
  int idx = blockIdx.x * blockDim.x + threadIdx.x;
  int total = (K * Nn) >> 1;
  if (idx >= total) return;
  int f = idx >> 8, r = idx & 255;
  int lane = r >> 3, v = r & 7;
  int Nt = Nn >> 4;
  int kt = f / Nt, nt = f % Nt;
  int n = (nt << 4) + (lane & 15);
  int k = (kt << 5) + ((lane >> 4) << 4) + (v << 1);
  Bp[idx] = packbf(W[(size_t)k * Nn + n], W[(size_t)(k + 1) * Nn + n]);
}

// v[k,h] = sum_c le[k, h*32+c] * a_e[h,c]   (factored edge projection, 64x8)
__global__ void compute_v_kernel(const float* __restrict__ le, const float* __restrict__ ae,
                                 float* __restrict__ v) {
  int i = blockIdx.x * blockDim.x + threadIdx.x;
  if (i >= 64 * 8) return;
  int k = i >> 3, h = i & 7;
  float acc = 0.f;
  #pragma unroll
  for (int c = 0; c < 32; ++c) acc += le[(size_t)k * 256 + h * 32 + c] * ae[h * 32 + c];
  v[k * 8 + h] = acc;
}

__global__ void cnt_kernel(const int* __restrict__ dst, float* __restrict__ cnt, int E) {
  int e = blockIdx.x * blockDim.x + threadIdx.x;
  if (e < E) atomicAdd(&cnt[dst[e]], 1.0f);
}

// --------------------------------------------------------- WMMA bf16 GEMM
// A: M x K row-major bf16 (as dwords, K/2 per row). Bp: packed fragments.
__global__ void gemm_wmma_bf16(const unsigned* __restrict__ Abf,
                               const unsigned* __restrict__ Bp,
                               const float* __restrict__ bias,
                               float* __restrict__ Cf,
                               unsigned short* __restrict__ Cbf,
                               int M, int K, int Nn, int relu) {
  const int Mt = M >> 4, Nt = Nn >> 4, Kd = K >> 5;
  const int wave = (blockIdx.x * blockDim.x + threadIdx.x) >> 5;
  const int lane = threadIdx.x & 31;
  if (wave >= Mt * Nt) return;                 // wave-uniform: EXEC stays all-ones
  const int mt = wave % Mt;
  const int nt = wave / Mt;                    // adjacent waves share B fragments
  const int hi = lane >> 4;
  const int ln = lane & 15;
  const int m  = (mt << 4) + ln;
  const u32x4* Arow = (const u32x4*)(Abf + (size_t)m * (K >> 1));
  v8f acc = {};
  for (int ks = 0; ks < Kd; ++ks) {
    ABFrag a, b;
    a.q[0] = Arow[ks * 4 + 2 * hi];
    a.q[1] = Arow[ks * 4 + 2 * hi + 1];
    __builtin_prefetch((const void*)(Arow + (ks + 1) * 4 + hi), 0, 0);
    const u32x4* Bf = (const u32x4*)(Bp + ((size_t)(ks * Nt + nt) << 8)) + (lane << 1);
    b.q[0] = Bf[0];
    b.q[1] = Bf[1];
    acc = __builtin_amdgcn_wmma_f32_16x16x32_bf16(false, a.v, false, b.v,
                                                  (short)0, acc, false, false);
  }
  const int col = (nt << 4) + ln;
  const float bb = bias ? bias[col] : 0.0f;
  #pragma unroll
  for (int v = 0; v < 8; ++v) {
    int row = (mt << 4) + v + (hi << 3);
    float x = acc[v] + bb;
    if (relu) x = fmaxf(x, 0.0f);
    if (Cf)  Cf [(size_t)row * Nn + col] = x;
    if (Cbf) Cbf[(size_t)row * Nn + col] = (unsigned short)f2bf1(x);
  }
}

// --------------------------------------------------------------- GAT parts
// a_src[n,h], a_dst[n,h] from xs
__global__ void attn_ab_kernel(const float* __restrict__ xs, const float* __restrict__ as,
                               const float* __restrict__ ad, float* __restrict__ asrc,
                               float* __restrict__ adst, int N) {
  int i = blockIdx.x * blockDim.x + threadIdx.x;
  if (i >= N * 8) return;
  int n = i >> 3, h = i & 7;
  const float* xr = xs + (size_t)n * 256 + h * 32;
  float s = 0.f, d = 0.f;
  #pragma unroll
  for (int c = 0; c < 32; ++c) { float x = xr[c]; s += x * as[h * 32 + c]; d += x * ad[h * 32 + c]; }
  asrc[i] = s; adst[i] = d;
}

// et[e,h] = ew[e]·v[:,h] ; accumulate per-dst sums (for self-loop mean term)
__global__ void edge_term_kernel(const int* __restrict__ dst, const float* __restrict__ ew,
                                 const float* __restrict__ v, float* __restrict__ et,
                                 float* __restrict__ lsum, int E) {
  int i = blockIdx.x * blockDim.x + threadIdx.x;
  if (i >= E * 8) return;
  int e = i >> 3, h = i & 7;
  const float* wr = ew + (size_t)e * 64;
  float acc = 0.f;
  #pragma unroll
  for (int k = 0; k < 64; ++k) acc += wr[k] * v[k * 8 + h];
  et[i] = acc;
  atomicAdd(&lsum[(size_t)dst[e] * 8 + h], acc);
}

__global__ void alpha_max_kernel(const int* __restrict__ src, const int* __restrict__ dst,
                                 const float* __restrict__ et, const float* __restrict__ lsum,
                                 const float* __restrict__ cnt, const float* __restrict__ asrc,
                                 const float* __restrict__ adst, float* __restrict__ alpha,
                                 float* __restrict__ m, int E, int N) {
  int i = blockIdx.x * blockDim.x + threadIdx.x;
  if (i >= (E + N) * 8) return;
  int e = i >> 3, h = i & 7;
  int s, d; float t;
  if (e < E) { s = src[e]; d = dst[e]; t = et[i]; }
  else       { s = d = e - E; t = lsum[(size_t)s * 8 + h] / fmaxf(cnt[s], 1.0f); }
  float a = asrc[(size_t)s * 8 + h] + adst[(size_t)d * 8 + h] + t;
  a = a > 0.f ? a : 0.2f * a;                 // leaky_relu(0.2)
  alpha[i] = a;
  atomicMaxF(&m[(size_t)d * 8 + h], a);
}

__global__ void exp_den_kernel(const int* __restrict__ dst, const float* __restrict__ m,
                               float* __restrict__ alpha, float* __restrict__ den, int E, int N) {
  int i = blockIdx.x * blockDim.x + threadIdx.x;
  if (i >= (E + N) * 8) return;
  int e = i >> 3, h = i & 7;
  int d = (e < E) ? dst[e] : (e - E);
  float ex = __expf(alpha[i] - m[(size_t)d * 8 + h]);
  alpha[i] = ex;
  atomicAdd(&den[(size_t)d * 8 + h], ex);
}

// one wave per (edge or self-loop): out[dst] += w * xs[src]
__global__ void aggregate_kernel(const int* __restrict__ src, const int* __restrict__ dst,
                                 const float* __restrict__ ex, const float* __restrict__ den,
                                 const float* __restrict__ xs, float* __restrict__ out,
                                 int E, int N) {
  int wid = (blockIdx.x * blockDim.x + threadIdx.x) >> 5;
  int lane = threadIdx.x & 31;
  if (wid >= E + N) return;
  int s, d;
  if (wid < E) { s = src[wid]; d = dst[wid]; } else { s = d = wid - E; }
  const float* exr = ex  + (size_t)wid * 8;
  const float* dr  = den + (size_t)d * 8;
  const float* xr  = xs  + (size_t)s * 256;
  float* orow      = out + (size_t)d * 256;
  #pragma unroll
  for (int j = 0; j < 8; ++j) {
    float w = exr[j] / (dr[j] + 1e-16f);
    atomicAdd(&orow[j * 32 + lane], w * xr[j * 32 + lane]);
  }
}

// LN (unbiased std, eps on std) + leaky(0.01) + residual; optional bias, optional bf16 mirror
__global__ void post_ln_kernel(const float* __restrict__ h, const float* __restrict__ bias,
                               const float* __restrict__ g, const float* __restrict__ be,
                               const float* __restrict__ nf_in, float* __restrict__ nf_out,
                               unsigned short* __restrict__ nf_bf, int N) {
  int row = (blockIdx.x * blockDim.x + threadIdx.x) >> 5;
  int lane = threadIdx.x & 31;
  if (row >= N) return;
  const float* hr = h + (size_t)row * 256;
  float x[8], s = 0.f;
  #pragma unroll
  for (int j = 0; j < 8; ++j) {
    int c = j * 32 + lane;
    float v = hr[c] + (bias ? bias[c] : 0.f);
    x[j] = v; s += v;
  }
  for (int o = 16; o; o >>= 1) s += __shfl_xor(s, o, 32);
  float mu = s * (1.0f / 256.0f);
  float q = 0.f;
  #pragma unroll
  for (int j = 0; j < 8; ++j) { float dlt = x[j] - mu; q += dlt * dlt; }
  for (int o = 16; o; o >>= 1) q += __shfl_xor(q, o, 32);
  float inv = 1.0f / (sqrtf(q * (1.0f / 255.0f)) + 1e-6f);
  #pragma unroll
  for (int j = 0; j < 8; ++j) {
    int c = j * 32 + lane;
    float y = g[c] * (x[j] - mu) * inv + be[c];
    y = y > 0.f ? y : 0.01f * y;
    float r = nf_in[(size_t)row * 256 + c] + y;
    nf_out[(size_t)row * 256 + c] = r;
    if (nf_bf) nf_bf[(size_t)row * 256 + c] = (unsigned short)f2bf1(r);
  }
}

// ------------------------------------------------------------------- launch
extern "C" void kernel_launch(void* const* d_in, const int* in_sizes, int n_in,
                              void* d_out, int out_size, void* d_ws, size_t ws_size,
                              hipStream_t stream) {
  const float* nf  = (const float*)d_in[0];
  const int*   ei  = (const int*)  d_in[1];
  const float* ew  = (const float*)d_in[2];
  const float* lin[2]   = { (const float*)d_in[3],  (const float*)d_in[9]  };
  const float* le[2]    = { (const float*)d_in[4],  (const float*)d_in[10] };
  const float* a_s[2]   = { (const float*)d_in[5],  (const float*)d_in[11] };
  const float* a_d[2]   = { (const float*)d_in[6],  (const float*)d_in[12] };
  const float* a_e[2]   = { (const float*)d_in[7],  (const float*)d_in[13] };
  const float* bsv[2]   = { (const float*)d_in[8],  (const float*)d_in[14] };
  const float* gamma[3] = { (const float*)d_in[15], (const float*)d_in[17], (const float*)d_in[19] };
  const float* beta[3]  = { (const float*)d_in[16], (const float*)d_in[18], (const float*)d_in[20] };
  const float* w1  = (const float*)d_in[21];
  const float* fb1 = (const float*)d_in[22];
  const float* w2  = (const float*)d_in[23];
  const float* fb2 = (const float*)d_in[24];

  const int N = in_sizes[0] / 256;
  const int E = in_sizes[1] / 2;
  const int* src = ei;
  const int* dst = ei + E;

  // workspace bump allocator (256B aligned)
  char* wsb = (char*)d_ws;
  size_t off = 0;
  auto alloc = [&](size_t bytes) -> void* {
    void* p = wsb + off;
    off = (off + bytes + 255) & ~(size_t)255;
    return p;
  };
  float*          nfA  = (float*)         alloc((size_t)N * 256 * 4);
  unsigned short* nfbf = (unsigned short*)alloc((size_t)N * 256 * 2);
  float*          xs   = (float*)         alloc((size_t)N * 256 * 4);
  float*          agg  = (float*)         alloc((size_t)N * 256 * 4);
  float*          asrc = (float*)         alloc((size_t)N * 8 * 4);
  float*          adst = (float*)         alloc((size_t)N * 8 * 4);
  float*          mbuf = (float*)         alloc((size_t)N * 8 * 4);
  float*          den  = (float*)         alloc((size_t)N * 8 * 4);
  float*          lsum = (float*)         alloc((size_t)N * 8 * 4);
  float*          cnt  = (float*)         alloc((size_t)N * 4);
  unsigned*       lp[2]; lp[0] = (unsigned*)alloc(256 * 256 * 2); lp[1] = (unsigned*)alloc(256 * 256 * 2);
  unsigned*       w1p  = (unsigned*)      alloc(256 * 1024 * 2);
  unsigned*       w2p  = (unsigned*)      alloc(1024 * 256 * 2);
  float*          vv[2]; vv[0] = (float*) alloc(64 * 8 * 4); vv[1] = (float*)alloc(64 * 8 * 4);
  float*          et    = (float*)        alloc((size_t)E * 8 * 4);
  float*          alpha = (float*)        alloc((size_t)(E + N) * 8 * 4);
  unsigned short* tbf   = (unsigned short*)et;   // FFN hidden (bf16) reuses edge buffers

  const int T = 256;
  auto nb = [](long long n, int t) { return (unsigned)((n + t - 1) / t); };

  // ---- prep (layer independent)
  hipMemcpyAsync(nfA, nf, (size_t)N * 256 * 4, hipMemcpyDeviceToDevice, stream);
  fill_kernel<<<nb(N, T), T, 0, stream>>>((unsigned*)cnt, 0u, N);
  cnt_kernel<<<nb(E, T), T, 0, stream>>>(dst, cnt, E);
  pack_b_kernel<<<nb(256 * 256 / 2, T), T, 0, stream>>>(lin[0], lp[0], 256, 256);
  pack_b_kernel<<<nb(256 * 256 / 2, T), T, 0, stream>>>(lin[1], lp[1], 256, 256);
  pack_b_kernel<<<nb(256 * 1024 / 2, T), T, 0, stream>>>(w1, w1p, 256, 1024);
  pack_b_kernel<<<nb(1024 * 256 / 2, T), T, 0, stream>>>(w2, w2p, 1024, 256);
  compute_v_kernel<<<nb(64 * 8, T), T, 0, stream>>>(le[0], a_e[0], vv[0]);
  compute_v_kernel<<<nb(64 * 8, T), T, 0, stream>>>(le[1], a_e[1], vv[1]);
  cvt_bf16_kernel<<<nb((size_t)N * 128, T), T, 0, stream>>>(nfA, (unsigned*)nfbf, N * 128);

  // ---- two GAT layers
  for (int L = 0; L < 2; ++L) {
    fill_kernel<<<nb(N * 8, T), T, 0, stream>>>((unsigned*)mbuf, 0xFF800000u, N * 8);
    fill_kernel<<<nb(N * 8, T), T, 0, stream>>>((unsigned*)den, 0u, N * 8);
    fill_kernel<<<nb(N * 8, T), T, 0, stream>>>((unsigned*)lsum, 0u, N * 8);
    fill_kernel<<<nb((size_t)N * 256, T), T, 0, stream>>>((unsigned*)agg, 0u, N * 256);

    edge_term_kernel<<<nb((size_t)E * 8, T), T, 0, stream>>>(dst, ew, vv[L], et, lsum, E);

    int tiles = (N / 16) * (256 / 16);
    gemm_wmma_bf16<<<nb((size_t)tiles * 32, 128), 128, 0, stream>>>(
        (const unsigned*)nfbf, lp[L], nullptr, xs, nullptr, N, 256, 256, 0);

    attn_ab_kernel<<<nb(N * 8, T), T, 0, stream>>>(xs, a_s[L], a_d[L], asrc, adst, N);
    alpha_max_kernel<<<nb((size_t)(E + N) * 8, T), T, 0, stream>>>(src, dst, et, lsum, cnt,
                                                                   asrc, adst, alpha, mbuf, E, N);
    exp_den_kernel<<<nb((size_t)(E + N) * 8, T), T, 0, stream>>>(dst, mbuf, alpha, den, E, N);
    aggregate_kernel<<<nb((size_t)(E + N) * 32, T), T, 0, stream>>>(src, dst, alpha, den, xs,
                                                                    agg, E, N);
    post_ln_kernel<<<nb((size_t)N * 32, T), T, 0, stream>>>(agg, bsv[L], gamma[L], beta[L],
                                                            nfA, nfA, nfbf, N);
  }

  // ---- FFN: relu(nf @ w1 + fb1) @ w2 + fb2, then LN + residual -> d_out
  int t1 = (N / 16) * (1024 / 16);
  gemm_wmma_bf16<<<nb((size_t)t1 * 32, 128), 128, 0, stream>>>(
      (const unsigned*)nfbf, w1p, fb1, nullptr, tbf, N, 256, 1024, 1);
  int t2 = (N / 16) * (256 / 16);
  gemm_wmma_bf16<<<nb((size_t)t2 * 32, 128), 128, 0, stream>>>(
      (const unsigned*)tbf, w2p, fb2, agg, nullptr, N, 1024, 256, 0);
  post_ln_kernel<<<nb((size_t)N * 32, T), T, 0, stream>>>(agg, nullptr, gamma[2], beta[2],
                                                          nfA, (float*)d_out, nullptr, N);
}
